// SpikingReservoirLoaded_12945031430483
// MI455X (gfx1250) — matrix operand order repaired and checked by reading, not compiled, not run on hardware
//
#include <hip/hip_runtime.h>

#define NRES   2048
#define BATCH  32
#define TSTEPS 500
#define THRESH 1.0f
#define BETA   0.95f

typedef __attribute__((ext_vector_type(16))) __bf16 v16bf;
typedef __attribute__((ext_vector_type(8)))  float  v8f;

// ---------------------------------------------------------------------------
// WMMA fragment layouts used here (wave32; A/B in contiguous K order, not the ISA table order):
//  B (32x16, K x N):  lane l: N=l&15; lanes 0-15 hold K=0..15, lanes 16-31 hold
//                     K=16..31; element i -> K = (l>>4)*16 + i.
//  A (16x32, M x K):  lane l<16: M=l,    K=0..15 (element i -> K = i)
//                     lane l>=16: M=l-16, K=16..31 (element i -> K = 16 + i)
//  C (16x16 f32):     lanes 0-15: N=lane, M=r; lanes 16-31: N=lane-16, M=8+r.
//
// W^T packed once per launch into B-fragment blobs
//   [nt(128)][kt(64)][lane(32)][elem(16)] bf16  (1 KB per 32x16 K-tile),
// as bf16 hi + bf16 residual lo: two K=32 bf16 WMMAs recover ~fp32 accuracy on
// the fast XDL pipe. 16 MB total -> resident in the 192 MB L2 across all steps.
//
// Spikes staged between steps already in A-fragment layout (exact: spk is {0,1}):
//   [mt(2)][kt(64)][lane(32)][elem(16)] bf16.
// ---------------------------------------------------------------------------

__global__ void pack_w_kernel(const float* __restrict__ W,
                              __bf16* __restrict__ whi,
                              __bf16* __restrict__ wlo, int use_lo) {
  int tid  = blockIdx.x * blockDim.x + threadIdx.x;   // 8192 tiles * 32 lanes
  int lane = tid & 31;
  int tile = tid >> 5;            // tile = nt*64 + kt
  int kt   = tile & 63;
  int nt   = tile >> 6;
  int n     = nt * 16 + (lane & 15);
  int kbase = kt * 32 + (lane >> 4) * 16;
  size_t base = (size_t)tile * 512 + (size_t)lane * 16;
  #pragma unroll
  for (int i = 0; i < 16; ++i) {
    float w = W[(size_t)n * NRES + (kbase + i)];      // Wt[k][n] == W[n][k]
    __bf16 hi = (__bf16)w;
    whi[base + i] = hi;
    if (use_lo) wlo[base + i] = (__bf16)(w - (float)hi);
  }
}

__global__ void zero_init_kernel(unsigned* __restrict__ apack0_u32, // 32768 dw
                                 unsigned* __restrict__ zmem_u32,   // 65536 dw
                                 unsigned* __restrict__ cnt) {
  int i = blockIdx.x * blockDim.x + threadIdx.x;
  if (i < (BATCH * NRES) / 2) apack0_u32[i] = 0u;
  if (i < (BATCH * NRES))     zmem_u32[i]   = 0u;
  if (i == 0) *cnt = 0u;
}

// ---------------------------------------------------------------------------
// One timestep. Grid: 256 blocks (one per 16x16 C-tile), 128 threads = 4 waves.
// Split-K: wave w handles K-blobs [w*16, w*16+16); LDS reduction; wave 0 runs
// the fused RLeaky epilogue. 1024 waves total for latency hiding; per-wave
// dependent chain is 16 double-buffered iterations of 2 WMMAs.
// ---------------------------------------------------------------------------
template <int USE_LO>
__global__ __launch_bounds__(128)
void step_kernel(const __bf16* __restrict__ apack_in,  // (2,64,32,16) bf16
                 const float* __restrict__ mem_prev,   // (32,2048)
                 const float* __restrict__ x,          // (B,T)
                 const float* __restrict__ bias,       // (N)
                 const __bf16* __restrict__ whi,
                 const __bf16* __restrict__ wlo,
                 int t,
                 float* __restrict__ spk_rec_t,        // (32,2048) in d_out
                 float* __restrict__ mem_rec_t,        // (32,2048) in d_out
                 __bf16* __restrict__ apack_out,       // next step's A stage
                 unsigned* __restrict__ cnt) {
  __shared__ float red[4][32][8];                      // 4 KB of the 320 KB LDS

  const int lane  = threadIdx.x & 31;
  const int w     = threadIdx.x >> 5;   // 0..3 : K-split
  const int nt    = blockIdx.x >> 1;    // 0..127
  const int mt    = blockIdx.x & 1;     // 0..1
  const int khalf = lane >> 4;
  const int l15   = lane & 15;

  const __bf16* atile = apack_in + (size_t)(mt * 64 + w * 16) * 512 + (size_t)lane * 16;
  const __bf16* btile = whi + (size_t)(nt * 64 + w * 16) * 512 + (size_t)lane * 16;
  const __bf16* ltile = wlo + (size_t)(nt * 64 + w * 16) * 512 + (size_t)lane * 16;

  // Register double-buffered K-loop: loads for iter i+1 in flight during the
  // two WMMAs of iter i.
  v8f acc = {};
  v16bf a  = *(const v16bf*)(atile);
  v16bf bh = *(const v16bf*)(btile);
  v16bf bl = {};
  if (USE_LO) bl = *(const v16bf*)(ltile);
  #pragma unroll
  for (int i = 0; i < 16; ++i) {
    const int j = (i + 1) & 15;                        // wraps; always valid
    v16bf a_n  = *(const v16bf*)(atile + (size_t)j * 512);
    v16bf bh_n = *(const v16bf*)(btile + (size_t)j * 512);
    v16bf bl_n = {};
    if (USE_LO) bl_n = *(const v16bf*)(ltile + (size_t)j * 512);
    if (i < 14) {
      __builtin_prefetch(btile + (size_t)(i + 2) * 512, 0, 3);
      if (USE_LO) __builtin_prefetch(ltile + (size_t)(i + 2) * 512, 0, 3);
    }
    acc = __builtin_amdgcn_wmma_f32_16x16x32_bf16(false, a, false, bh,
                                                  (short)0, acc, false, false);
    if (USE_LO)
      acc = __builtin_amdgcn_wmma_f32_16x16x32_bf16(false, a, false, bl,
                                                    (short)0, acc, false, false);
    a = a_n; bh = bh_n; bl = bl_n;
  }

  // Cross-wave split-K reduction through LDS.
  #pragma unroll
  for (int r = 0; r < 8; ++r) red[w][lane][r] = acc[r];
  __syncthreads();
  if (w != 0) return;

  #pragma unroll
  for (int r = 0; r < 8; ++r)
    acc[r] = red[0][lane][r] + red[1][lane][r] + red[2][lane][r] + red[3][lane][r];

  // Fused RLeaky pointwise epilogue on the C-tile (wave 0 only).
  const int n_idx = nt * 16 + l15;
  const float bval = bias[n_idx];
  // This lane's column position in next step's A-fragment (fixed per lane):
  const int kt_o   = n_idx >> 5;
  const int kk     = n_idx & 31;
  const int lhalf  = kk >> 4;                         // -> A lanes 16-31
  const int elem_o = kk & 15;
  __bf16* aout = apack_out + (size_t)(mt * 64 + kt_o) * 512
                           + (size_t)(lhalf * 16) * 16 + elem_o;

  unsigned local = 0;
  #pragma unroll
  for (int r = 0; r < 8; ++r) {
    int m       = khalf ? (8 + r) : r;                // row within batch tile
    int b_idx   = mt * 16 + m;
    size_t o    = (size_t)b_idx * NRES + n_idx;
    float memp  = mem_prev[o];
    float reset = ((memp - THRESH) > 0.0f) ? 1.0f : 0.0f;
    float base  = BETA * memp + x[(size_t)b_idx * TSTEPS + t] + acc[r] + bval;
    float memn  = base * (1.0f - reset);
    float spkn  = ((memn - THRESH) > 0.0f) ? 1.0f : 0.0f;
    mem_rec_t[o] = memn;
    spk_rec_t[o] = spkn;
    aout[(size_t)m * 16] = (__bf16)spkn;              // scatter into A-lane m
    local += (spkn != 0.0f) ? 1u : 0u;
  }
  #pragma unroll
  for (int off = 16; off > 0; off >>= 1) local += __shfl_down(local, off, 32);
  if (lane == 0) atomicAdd(cnt, local);
}

__global__ void finalize_kernel(const unsigned* __restrict__ cnt,
                                float* __restrict__ out0) {
  *out0 = (float)(*cnt) * (1.0f / (float)((size_t)TSTEPS * BATCH * NRES));
}

// ---------------------------------------------------------------------------
extern "C" void kernel_launch(void* const* d_in, const int* in_sizes, int n_in,
                              void* d_out, int out_size, void* d_ws, size_t ws_size,
                              hipStream_t stream) {
  (void)in_sizes; (void)n_in; (void)out_size;
  const float* x    = (const float*)d_in[0];   // (32,500,1)
  const float* W    = (const float*)d_in[1];   // (2048,2048)
  const float* bias = (const float*)d_in[2];   // (2048,)

  float* out     = (float*)d_out;
  float* spk_rec = out + 1;
  float* mem_rec = out + 1 + (size_t)TSTEPS * BATCH * NRES;

  const size_t packBytes  = (size_t)NRES * NRES * sizeof(__bf16);   // 8 MB each
  const size_t stageElems = (size_t)BATCH * NRES;                   // 64K
  const size_t stageBytes = stageElems * sizeof(__bf16);            // 128 KB
  const size_t zmemBytes  = stageElems * sizeof(float);             // 256 KB
  const size_t needFull   = 2 * packBytes + 2 * stageBytes + zmemBytes + 64;
  const int use_lo = (ws_size >= needFull) ? 1 : 0;

  char* p = (char*)d_ws;
  __bf16* whi = (__bf16*)p;  p += packBytes;
  __bf16* wlo = whi;
  if (use_lo) { wlo = (__bf16*)p; p += packBytes; }
  __bf16* apack0 = (__bf16*)p; p += stageBytes;
  __bf16* apack1 = (__bf16*)p; p += stageBytes;
  float*  zmem   = (float*)p;  p += zmemBytes;    // dedicated f32 zeros for t=0
  unsigned* cnt  = (unsigned*)p;
  __bf16* stages[2] = { apack0, apack1 };

  // Per-call init (deterministic; harness does not re-poison between replays).
  zero_init_kernel<<<(int)((stageElems + 255) / 256), 256, 0, stream>>>(
      (unsigned*)apack0, (unsigned*)zmem, cnt);
  // Repack W^T into bf16 hi/lo B-fragments (resident in L2 afterwards).
  pack_w_kernel<<<(8192 * 32) / 256, 256, 0, stream>>>(W, whi, wlo, use_lo);

  for (int t = 0; t < TSTEPS; ++t) {
    const __bf16* a_in = stages[t & 1];                  // t=0 -> zeros
    __bf16* a_out      = stages[(t + 1) & 1];
    const float* memp  = (t == 0) ? zmem
                                  : (mem_rec + (size_t)(t - 1) * stageElems);
    float* spk_t = spk_rec + (size_t)t * stageElems;
    float* mem_t = mem_rec + (size_t)t * stageElems;
    if (use_lo)
      step_kernel<1><<<256, 128, 0, stream>>>(a_in, memp, x, bias, whi, wlo,
                                              t, spk_t, mem_t, a_out, cnt);
    else
      step_kernel<0><<<256, 128, 0, stream>>>(a_in, memp, x, bias, whi, wlo,
                                              t, spk_t, mem_t, a_out, cnt);
  }
  finalize_kernel<<<1, 1, 0, stream>>>(cnt, out);
}
